// SREChead_91079076479600
// MI455X (gfx1250) — matrix operand order repaired and checked by reading, not compiled, not run on hardware
//
#include <hip/hip_runtime.h>
#include <hip/hip_bf16.h>

// ---------- types ----------
typedef __bf16  v16bf __attribute__((ext_vector_type(16)));
typedef float   v8f   __attribute__((ext_vector_type(8)));
typedef unsigned int v4u __attribute__((ext_vector_type(4)));
typedef int     v8i   __attribute__((ext_vector_type(8)));
typedef int     v4i   __attribute__((ext_vector_type(4)));

#if defined(__has_builtin)
#if __has_builtin(__builtin_amdgcn_tensor_load_to_lds)
#define USE_TDM 1
#endif
#endif

#define BATCH 64
#define HGT 80
#define WID 80

__device__ __forceinline__ unsigned short f2bf(float f) {
    unsigned u = __builtin_bit_cast(unsigned, f);
    u += 0x7FFFu + ((u >> 16) & 1u);           // round-to-nearest-even
    return (unsigned short)(u >> 16);
}
__device__ __forceinline__ float bf2f(unsigned short h) {
    return __builtin_bit_cast(float, (unsigned)h << 16);
}

struct B32B { uint4 a, b; };
__device__ __forceinline__ v16bf ld_frag(const unsigned short* p) {
    B32B r;
    r.a = *(const uint4*)(p);
    r.b = *(const uint4*)(p + 8);
    return __builtin_bit_cast(v16bf, r);
}

// ---------- weight packing into A-fragment layout (16-bit A 16x32, contiguous k = 16*hi + j, not the ISA table order) ----------
// packed layout: [tile(4)][kchunk][lane(32)][j(16 halves)], K ordering k = kpos*CIN + ic
__global__ __launch_bounds__(256) void pack_weights(const float* __restrict__ w,
                                                    unsigned short* __restrict__ pk,
                                                    int CIN, int KS) {
    const int NCHUNK = CIN * KS * KS / 32;
    const int total = 4 * NCHUNK * 512;
    int idx = blockIdx.x * 256 + threadIdx.x;
    if (idx >= total) return;
    int j    = idx & 15;
    int lane = (idx >> 4) & 31;
    int kc   = (idx >> 9) % NCHUNK;
    int tile = idx / (NCHUNK * 512);
    int hi = lane >> 4, m = lane & 15, v = j >> 1, pos = j & 1;
    // lanes 0-15 hold K 0..15; lanes 16-31 hold K 16..31
    int kin = hi * 16 + v * 2 + pos;
    int kg  = kc * 32 + kin;
    int kpos = kg / CIN, ic = kg % CIN;
    int oc = tile * 16 + m;
    float val = w[(oc * CIN + ic) * (KS * KS) + kpos];
    pk[idx] = f2bf(val);
}

// ---------- fused conv + BN-affine + SiLU, implicit GEMM via WMMA ----------
// Output: NHWC bf16.  Block = 256 thr = 8 waves; tile = 8 rows x 16 cols of pixels;
// wave w handles row w (16 pixels, N-tile) for all 4 oc tiles (M).
template <int CIN, int KS, bool IN_NCHW_F32>
__global__ __launch_bounds__(256) void conv_wmma(const void* __restrict__ in_,
                                                 const unsigned short* __restrict__ pw,
                                                 const float* __restrict__ scale,
                                                 const float* __restrict__ shift,
                                                 unsigned short* __restrict__ out) {
    constexpr int TH = 8, TW = 16;
    constexpr int PAD = (KS - 1) / 2;
    constexpr int PH = TH + KS - 1, PW = TW + KS - 1;
    constexpr int NCHUNK = CIN * KS * KS / 32;
    constexpr int NWH = 4 * NCHUNK * 32 * 16;         // packed-weight halves

    __shared__ alignas(16) unsigned short lds_w[NWH];
    __shared__ alignas(16) unsigned short lds_p[PH * PW * CIN];

    const int tid = threadIdx.x;
    const int lane = tid & 31;
    const int wrow = tid >> 5;                        // wave id == row in tile
    const int x0 = blockIdx.x * TW, y0 = blockIdx.y * TH;
    const int b = blockIdx.z;

    // ---- packed weights -> LDS via Tensor Data Mover (1-D tile) ----
#ifdef USE_TDM
    if (wrow == 0) {
        unsigned lds_off = (unsigned)(size_t)(&lds_w[0]);
        unsigned long long ga = (unsigned long long)(size_t)pw;
        v4u g0;
        g0[0] = 1u;                                   // count=1, no gather
        g0[1] = lds_off;                              // lds_addr
        g0[2] = (unsigned)ga;                         // global_addr[31:0]
        g0[3] = (unsigned)((ga >> 32) & 0x01FFFFFFu) | (2u << 30);  // addr[56:32] | type=2
        const unsigned nw = (unsigned)NWH;
        v8i g1;
        g1[0] = 0x10000;                              // data_size=1 (2 bytes), mask=0
        g1[1] = (int)((nw & 0xFFFFu) << 16);          // tensor_dim0[15:0]
        g1[2] = (int)(((nw >> 16) & 0xFFFFu) | (1u << 16)); // tensor_dim0[31:16], tensor_dim1=1
        g1[3] = (int)((nw & 0xFFFFu) << 16);          // tile_dim0 = nw
        g1[4] = 0;                                    // tile_dim1/2 unused
        g1[5] = (int)nw;                              // tensor_dim0_stride lo
        g1[6] = 0;
        g1[7] = 0;
        v4i z4 = {0, 0, 0, 0};
#if __clang_major__ >= 23
        v8i z8 = {0, 0, 0, 0, 0, 0, 0, 0};
        __builtin_amdgcn_tensor_load_to_lds(g0, g1, z4, z4, z8, 0);
#else
        __builtin_amdgcn_tensor_load_to_lds(g0, g1, z4, z4, 0);
#endif
        __builtin_amdgcn_s_wait_tensorcnt(0);
    }
#else
    for (int i = tid; i < NWH / 8; i += 256)
        ((uint4*)lds_w)[i] = ((const uint4*)pw)[i];
#endif

    // ---- input patch -> LDS (NHWC bf16), zero-padded halo ----
    if constexpr (IN_NCHW_F32) {
        // stem: input is x in NCHW f32; KS==1 so no halo, always in-bounds
        const float* in = (const float*)in_;
        constexpr int PATCH = PH * PW;
        for (int idx = tid; idx < PATCH * CIN; idx += 256) {
            int c = idx / PATCH, p = idx % PATCH;
            int gy = y0 + p / PW, gx = x0 + p % PW;
            float v = in[((size_t)(b * CIN + c) * (HGT * WID)) + gy * WID + gx];
            lds_p[p * CIN + c] = f2bf(v);
        }
    } else {
        const unsigned short* in = (const unsigned short*)in_;
        constexpr int CG = CIN / 8;
        for (int idx = tid; idx < PH * PW * CG; idx += 256) {
            int p = idx / CG, cg = idx % CG;
            int gy = y0 + p / PW - PAD, gx = x0 + p % PW - PAD;
            uint4 v = make_uint4(0u, 0u, 0u, 0u);
            if (gy >= 0 && gy < HGT && gx >= 0 && gx < WID)
                v = *(const uint4*)(in + ((size_t)(b * (HGT * WID) + gy * WID + gx) * CIN + cg * 8));
            *(uint4*)(&lds_p[p * CIN + cg * 8]) = v;
        }
    }
    __syncthreads();

    // ---- K loop: 4 WMMA per chunk (oc tiles), B frag from LDS patch ----
    v8f acc[4];
#pragma unroll
    for (int t = 0; t < 4; ++t)
#pragma unroll
        for (int v = 0; v < 8; ++v) acc[t][v] = 0.0f;

    const int pcol = lane & 15;
    const int hi = lane >> 4;
    for (int kc = 0; kc < NCHUNK; ++kc) {
        const int kpos = (kc * 32) / CIN;
        const int icb = (kc * 32) % CIN;
        const int dy = kpos / KS, dx = kpos % KS;
        // B 32x16: lanes 0-15 -> K 0..15 (ic icb..), lanes 16-31 -> K 16..31
        const int poff = ((wrow + dy) * PW + (pcol + dx)) * CIN + icb + hi * 16;
        v16bf bfrag = ld_frag(&lds_p[poff]);
#pragma unroll
        for (int t = 0; t < 4; ++t) {
            v16bf afrag = ld_frag(&lds_w[((t * NCHUNK + kc) * 32 + lane) * 16]);
            acc[t] = __builtin_amdgcn_wmma_f32_16x16x32_bf16(
                false, afrag, false, bfrag, (short)0, acc[t], false, false);
        }
    }

    // ---- epilogue: BN affine + SiLU, bf16 pack, NHWC store ----
    const int gy = y0 + wrow, gx = x0 + pcol;
    const size_t obase = ((size_t)(b * (HGT * WID) + gy * WID + gx)) * 64;
#pragma unroll
    for (int t = 0; t < 4; ++t) {
        const int mb = t * 16 + hi * 8;               // C/D: lanes 0-15 hold M 0..7, 16-31 hold M 8..15
        float4 s0 = *(const float4*)(scale + mb);
        float4 s1 = *(const float4*)(scale + mb + 4);
        float4 h0 = *(const float4*)(shift + mb);
        float4 h1 = *(const float4*)(shift + mb + 4);
        float sc[8] = {s0.x, s0.y, s0.z, s0.w, s1.x, s1.y, s1.z, s1.w};
        float sh[8] = {h0.x, h0.y, h0.z, h0.w, h1.x, h1.y, h1.z, h1.w};
        unsigned r[4];
#pragma unroll
        for (int v = 0; v < 4; ++v) {
            float f0 = acc[t][2 * v] * sc[2 * v] + sh[2 * v];
            float f1 = acc[t][2 * v + 1] * sc[2 * v + 1] + sh[2 * v + 1];
            f0 = f0 / (1.0f + __expf(-f0));           // SiLU
            f1 = f1 / (1.0f + __expf(-f1));
            r[v] = (unsigned)f2bf(f0) | ((unsigned)f2bf(f1) << 16);
        }
        *(uint4*)(out + obase + mb) = make_uint4(r[0], r[1], r[2], r[3]);
    }
}

// ---------- heads (4+1 channel 1x1) + YOLOX decode ----------
__global__ __launch_bounds__(256) void head_decode(const unsigned short* __restrict__ f,
                                                   const float* __restrict__ reg_w,
                                                   const float* __restrict__ reg_b,
                                                   const float* __restrict__ obj_w,
                                                   const float* __restrict__ obj_b,
                                                   float* __restrict__ out) {
    int idx = blockIdx.x * 256 + threadIdx.x;
    if (idx >= BATCH * HGT * WID) return;
    int p = idx % (HGT * WID);
    const unsigned short* fp = f + (size_t)idx * 64;
    float a0 = reg_b[0], a1 = reg_b[1], a2 = reg_b[2], a3 = reg_b[3], ao = obj_b[0];
#pragma unroll 8
    for (int c = 0; c < 64; ++c) {
        float v = bf2f(fp[c]);
        a0 += v * reg_w[c];
        a1 += v * reg_w[64 + c];
        a2 += v * reg_w[128 + c];
        a3 += v * reg_w[192 + c];
        ao += v * obj_w[c];
    }
    float gx = (float)(p % WID), gy = (float)(p / WID);
    float* o = out + (size_t)idx * 5;
    o[0] = (a0 + gx) * 32.0f;
    o[1] = (a1 + gy) * 32.0f;
    o[2] = __expf(a2) * 32.0f;
    o[3] = __expf(a3) * 32.0f;
    o[4] = 1.0f / (1.0f + __expf(-ao));
}

// ---------- launcher ----------
extern "C" void kernel_launch(void* const* d_in, const int* in_sizes, int n_in,
                              void* d_out, int out_size, void* d_ws, size_t ws_size,
                              hipStream_t stream) {
    const float* x          = (const float*)d_in[0];
    const float* stem_w     = (const float*)d_in[1];
    const float* stem_scale = (const float*)d_in[2];
    const float* stem_shift = (const float*)d_in[3];
    const float* c1_w       = (const float*)d_in[4];
    const float* c1_scale   = (const float*)d_in[5];
    const float* c1_shift   = (const float*)d_in[6];
    const float* c2_w       = (const float*)d_in[7];
    const float* c2_scale   = (const float*)d_in[8];
    const float* c2_shift   = (const float*)d_in[9];
    const float* reg_w      = (const float*)d_in[10];
    const float* reg_b      = (const float*)d_in[11];
    const float* obj_w      = (const float*)d_in[12];
    const float* obj_b      = (const float*)d_in[13];
    float* out = (float*)d_out;

    // workspace layout (bytes)
    char* ws = (char*)d_ws;
    const size_t PW_STEM = 0;                          // 4*4*512*2      = 16 KB
    const size_t PW_C1   = PW_STEM + 16384;            // 4*18*512*2     = 72 KB
    const size_t PW_C2   = PW_C1 + 73728;
    const size_t F0      = PW_C2 + 73728;              // 64*6400*64*2   = 50 MB
    const size_t F1      = F0 + (size_t)BATCH * HGT * WID * 64 * 2;
    unsigned short* pw_stem = (unsigned short*)(ws + PW_STEM);
    unsigned short* pw_c1   = (unsigned short*)(ws + PW_C1);
    unsigned short* pw_c2   = (unsigned short*)(ws + PW_C2);
    unsigned short* f0      = (unsigned short*)(ws + F0);
    unsigned short* f1      = (unsigned short*)(ws + F1);

    // 1) pack conv weights into A-fragment layout (bf16)
    pack_weights<<<(4 * 4 * 512 + 255) / 256, 256, 0, stream>>>(stem_w, pw_stem, 128, 1);
    pack_weights<<<(4 * 18 * 512 + 255) / 256, 256, 0, stream>>>(c1_w, pw_c1, 64, 3);
    pack_weights<<<(4 * 18 * 512 + 255) / 256, 256, 0, stream>>>(c2_w, pw_c2, 64, 3);

    // 2) conv pipeline: stem (NCHW f32 -> NHWC bf16), then two 3x3 convs
    dim3 grid(WID / 16, HGT / 8, BATCH);
    conv_wmma<128, 1, true ><<<grid, 256, 0, stream>>>(x,  pw_stem, stem_scale, stem_shift, f0);
    conv_wmma< 64, 3, false><<<grid, 256, 0, stream>>>(f0, pw_c1,   c1_scale,   c1_shift,   f1);
    conv_wmma< 64, 3, false><<<grid, 256, 0, stream>>>(f1, pw_c2,   c2_scale,   c2_shift,   f0);

    // 3) heads + decode
    head_decode<<<(BATCH * HGT * WID + 255) / 256, 256, 0, stream>>>(
        f0, reg_w, reg_b, obj_w, obj_b, out);
}
